// Encoder_Attention_74929999446608
// MI455X (gfx1250) — compile-verified, matrix operand order checked by reading, not run on hardware
//
#include <hip/hip_runtime.h>
#include <hip/hip_bf16.h>

#define BDIM 8
#define CDIM 128
#define HDIM 64
#define NPTS 4096
#define KNN  16
#define EPSV 1e-5f

typedef __attribute__((ext_vector_type(16))) __bf16 v16bf;
typedef __attribute__((ext_vector_type(8)))  float  v8f;

static __device__ inline v8f wmma_bf16(v16bf a, v16bf b, v8f c) {
  return __builtin_amdgcn_wmma_f32_16x16x32_bf16(false, a, false, b, (short)0, c,
                                                 false, false);
}

// Within a 32-wide K chunk: element e of lane's fragment -> local k index.
static __device__ inline int klocal_of(int e, int lane) {
  int koff = (lane & 16) ? 8 : 0;
  return (e < 8) ? (koff + e) : (16 + koff + (e - 8));
}

// A fragment from pre-packed bf16 fragment-major weights: 32 contiguous bytes/lane.
static __device__ inline v16bf fragA_pk(const __bf16* __restrict__ Wp, int frag, int lane) {
  return *reinterpret_cast<const v16bf*>(Wp + ((size_t)frag * 32 + lane) * 16);
}

// B fragment from a point-major f32 row (lane's own row, 8+8 consecutive floats).
static __device__ inline v16bf fragB_rowf32(const float* __restrict__ rowbase, int lane) {
  int koff = (lane & 16) ? 8 : 0;
  const float* p = rowbase + koff;
  v16bf f;
#pragma unroll
  for (int i = 0; i < 8; ++i) f[i] = (__bf16)p[i];
#pragma unroll
  for (int i = 0; i < 8; ++i) f[8 + i] = (__bf16)p[16 + i];
  return f;
}

// ------------------------------------------------------------- weight packing
// Row-major f32 [M][Kd] -> bf16 fragment-major: frag f = mt*(Kd/32)+kc,
// element t = f*512 + lane*16 + e.
__global__ void pack_weights(const float* __restrict__ src, __bf16* __restrict__ dst,
                             int M, int Kd) {
  int total = M * Kd;
  int nkc = Kd >> 5;
  for (int t = blockIdx.x * blockDim.x + threadIdx.x; t < total;
       t += gridDim.x * blockDim.x) {
    int f = t >> 9;
    int r = t & 511;
    int lane = r >> 4;
    int e = r & 15;
    int mt = f / nkc, kc = f - mt * nkc;
    int m = mt * 16 + (lane & 15);
    int k = kc * 32 + klocal_of(e, lane);
    dst[t] = (__bf16)src[(size_t)m * Kd + k];
  }
}

// ----------------------------------------------------------------- transpose
// [B][C][N] f32 -> [B][N][C] f32, 64-point tiles.
__global__ void transpose_bcn(const float* __restrict__ src, float* __restrict__ dst) {
  __shared__ float s[CDIM][64];
  int b = blockIdx.y, n0 = blockIdx.x * 64;
  int tx = threadIdx.x & 63, ty = threadIdx.x >> 6;       // 256 thr
  const float* sb = src + (size_t)b * CDIM * NPTS;
  for (int c = ty; c < CDIM; c += 4)
    s[c][tx] = sb[(size_t)c * NPTS + n0 + tx];
  __syncthreads();
  int p = threadIdx.x >> 2, qd = threadIdx.x & 3;         // 4 threads/point
  float* db = dst + ((size_t)b * NPTS + n0 + p) * CDIM + qd * 32;
#pragma unroll
  for (int i = 0; i < 32; ++i) db[i] = s[qd * 32 + i][p];
}

// ---------------------------------------------------------------- zero stats
__global__ void zero_stats(float* s) {
  if (threadIdx.x < 128) s[threadIdx.x] = 0.f;
}

// ---------------------------------------------------------------- QKV GEMMs
// grid (N/16, B, 3), block 256 (8 waves = 8 M-tiles). No LDS: B frags straight
// from feats_t rows; vectorized float4 stores to point-major q/k/v.
__global__ void qkv_kernel(const float* __restrict__ feats_t,
                           const __bf16* __restrict__ pWq, const float* bq,
                           const __bf16* __restrict__ pWk, const float* bk,
                           const __bf16* __restrict__ pWv, const float* bv,
                           float* qo, float* ko, float* vo) {
  int lane = threadIdx.x & 31;
  int wv   = threadIdx.x >> 5;
  int n0 = blockIdx.x * 16, b = blockIdx.y, which = blockIdx.z;
  const __bf16* W  = which == 0 ? pWq : (which == 1 ? pWk : pWv);
  const float*  bb = which == 0 ? bq  : (which == 1 ? bk  : bv);
  float*        out = which == 0 ? qo : (which == 1 ? ko : vo);
  int jj = lane & 15, mb = (lane & 16) ? 8 : 0;

  const float* frow = feats_t + ((size_t)b * NPTS + n0 + jj) * CDIM;
  v16bf bf[4];
#pragma unroll
  for (int kc = 0; kc < 4; ++kc) bf[kc] = fragB_rowf32(frow + kc * 32, lane);

  v8f acc = {};
#pragma unroll
  for (int kc = 0; kc < 4; ++kc)
    acc = wmma_bf16(fragA_pk(W, wv * 4 + kc, lane), bf[kc], acc);

  int c0 = wv * 16 + mb;
  union { float f[8]; float4 v4[2]; } o;
#pragma unroll
  for (int r = 0; r < 8; ++r) o.f[r] = acc[r] + bb[c0 + r];
  float* orow = out + ((size_t)b * NPTS + n0 + jj) * CDIM + c0;
  *(float4*)(orow)     = o.v4[0];
  *(float4*)(orow + 4) = o.v4[1];
}

// ---------------------------------------------------------------- brute KNN
__global__ void knn_kernel(const float* __restrict__ pts, int* __restrict__ knn) {
  __shared__ float sx[256], sy[256], sz[256];
  int b = blockIdx.y;
  int n = blockIdx.x * 256 + threadIdx.x;
  const float* P = pts + (size_t)b * 3 * NPTS;
  float qx = P[n], qy = P[NPTS + n], qz = P[2 * NPTS + n];

  float best[KNN]; int bid[KNN];
#pragma unroll
  for (int i = 0; i < KNN; ++i) { best[i] = 3.4e38f; bid[i] = 0; }

  for (int t = 0; t < NPTS; t += 256) {
    __syncthreads();
    sx[threadIdx.x] = P[t + threadIdx.x];
    sy[threadIdx.x] = P[NPTS + t + threadIdx.x];
    sz[threadIdx.x] = P[2 * NPTS + t + threadIdx.x];
    __syncthreads();
    for (int j = 0; j < 256; ++j) {
      float dx = qx - sx[j], dy = qy - sy[j], dz = qz - sz[j];
      float d  = dx * dx + dy * dy + dz * dz;
      if (d < best[KNN - 1]) {
        float dd = d; int jjj = t + j;
#pragma unroll
        for (int i = 0; i < KNN; ++i) {
          if (dd < best[i]) {
            float tb = best[i]; int ti = bid[i];
            best[i] = dd; bid[i] = jjj;
            dd = tb; jjj = ti;
          }
        }
      }
    }
  }
#pragma unroll
  for (int i = 0; i < KNN; ++i)
    knn[((size_t)b * NPTS + n) * KNN + i] = bid[i];
}

// ------------------------------------------------ pass A: BN statistics
// One point per wave, zero LDS in the MLP chain (register-local D->B handoff),
// shuffle pre-reduction before LDS/global atomics.
__global__ void stats_kernel(const float* __restrict__ pts,
                             const float* __restrict__ q_t, const float* __restrict__ k_t,
                             const int* __restrict__ knn,
                             const float* geo_W1, const float* geo_b1,
                             const __bf16* __restrict__ pGeoW2, const float* geo_b2,
                             const __bf16* __restrict__ pRelW1,
                             float* stat_sum, float* stat_sq) {
  __shared__ float s_sum[HDIM], s_sq[HDIM];
  int lane = threadIdx.x & 31, w = threadIdx.x >> 5;
  int pid = blockIdx.x * 8 + w;
  int b = pid >> 12, n = pid & (NPTS - 1);
  int jj = lane & 15, mb = (lane & 16) ? 8 : 0;
  if (threadIdx.x < HDIM) { s_sum[threadIdx.x] = 0.f; s_sq[threadIdx.x] = 0.f; }
  __syncthreads();

  int idxj = knn[(size_t)pid * KNN + jj];
  const float* P = pts + (size_t)b * 3 * NPTS;
  float rg0 = P[n] - P[idxj];
  float rg1 = P[NPTS + n] - P[NPTS + idxj];
  float rg2 = P[2 * NPTS + n] - P[2 * NPTS + idxj];

  // geo layer1 (3->64) straight into B fragments
  v16bf g1f[2];
#pragma unroll
  for (int ck = 0; ck < 2; ++ck)
#pragma unroll
    for (int e = 0; e < 16; ++e) {
      int h = ck * 32 + klocal_of(e, lane);
      float a = geo_b1[h] + geo_W1[h*3]*rg0 + geo_W1[h*3+1]*rg1 + geo_W1[h*3+2]*rg2;
      g1f[ck][e] = (__bf16)fmaxf(a, 0.f);
    }

  // geo layer2 fused into rel = q - knn_k + geo_emb (register-local)
  const float* qrow = q_t + ((size_t)b * NPTS + n) * CDIM;
  const float* krow = k_t + ((size_t)b * NPTS + idxj) * CDIM;
  v16bf relf[4];
#pragma unroll
  for (int mt = 0; mt < 8; ++mt) {
    v8f acc = {};
    acc = wmma_bf16(fragA_pk(pGeoW2, mt * 2 + 0, lane), g1f[0], acc);
    acc = wmma_bf16(fragA_pk(pGeoW2, mt * 2 + 1, lane), g1f[1], acc);
    int ck = mt >> 1, eo = (mt & 1) * 8, c0 = mt * 16 + mb;
#pragma unroll
    for (int r = 0; r < 8; ++r)
      relf[ck][eo + r] = (__bf16)(qrow[c0 + r] - krow[c0 + r] + acc[r] + geo_b2[c0 + r]);
  }

  // hpre (128->64); rel_b1 omitted (BN is bias-invariant). Shuffle-reduce over
  // the 16 neighbor columns, then 1 lane/half does the LDS atomics.
#pragma unroll
  for (int ht = 0; ht < 4; ++ht) {
    v8f acc = {};
#pragma unroll
    for (int kc = 0; kc < 4; ++kc)
      acc = wmma_bf16(fragA_pk(pRelW1, ht * 4 + kc, lane), relf[kc], acc);
    int c0 = ht * 16 + mb;
#pragma unroll
    for (int r = 0; r < 8; ++r) {
      float v = acc[r], v2 = v * v;
#pragma unroll
      for (int s = 1; s < 16; s <<= 1) {
        v  += __shfl_xor(v,  s, 32);
        v2 += __shfl_xor(v2, s, 32);
      }
      if (jj == 0) { atomicAdd(&s_sum[c0 + r], v); atomicAdd(&s_sq[c0 + r], v2); }
    }
  }
  __syncthreads();
  if (threadIdx.x < HDIM) {
    atomicAdd(&stat_sum[threadIdx.x], s_sum[threadIdx.x]);
    atomicAdd(&stat_sq[threadIdx.x],  s_sq[threadIdx.x]);
  }
}

// ---------------------------------------------------------------- BN finalize
__global__ void bn_finalize(const float* stat_sum, const float* stat_sq,
                            const float* g, const float* beta,
                            float* scale, float* shift) {
  int t = threadIdx.x;
  if (t < HDIM) {
    const float inv = 1.0f / (float)((size_t)BDIM * NPTS * KNN);
    float m  = stat_sum[t] * inv;
    float vv = stat_sq[t] * inv - m * m;
    float sc = g[t] * rsqrtf(vv + EPSV);
    scale[t] = sc;
    shift[t] = beta[t] - m * sc;
  }
}

// ------------------------------------ pass B: attention + aggregation (fused)
// One point per wave; entire chain register-local; softmax + aggregation via
// __shfl_xor butterflies over the 16 neighbor lanes. No LDS at all.
__global__ void attn_agg_kernel(const float* __restrict__ pts, const float* __restrict__ feats_t,
                                const float* __restrict__ q_t, const float* __restrict__ k_t,
                                const float* __restrict__ v_t, const int* __restrict__ knn,
                                const float* geo_W1, const float* geo_b1,
                                const __bf16* __restrict__ pGeoW2, const float* geo_b2,
                                const __bf16* __restrict__ pRelW1,
                                const __bf16* __restrict__ pRelW2, const float* rel_b2,
                                const float* bn_scale, const float* bn_shift,
                                float* __restrict__ agg_t) {
  int lane = threadIdx.x & 31, w = threadIdx.x >> 5;
  int pid = blockIdx.x * 8 + w;
  int b = pid >> 12, n = pid & (NPTS - 1);
  int jj = lane & 15, mb = (lane & 16) ? 8 : 0;

  int idxj = knn[(size_t)pid * KNN + jj];
  const float* P = pts + (size_t)b * 3 * NPTS;
  float rg0 = P[n] - P[idxj];
  float rg1 = P[NPTS + n] - P[NPTS + idxj];
  float rg2 = P[2 * NPTS + n] - P[2 * NPTS + idxj];

  v16bf g1f[2];
#pragma unroll
  for (int ck = 0; ck < 2; ++ck)
#pragma unroll
    for (int e = 0; e < 16; ++e) {
      int h = ck * 32 + klocal_of(e, lane);
      float a = geo_b1[h] + geo_W1[h*3]*rg0 + geo_W1[h*3+1]*rg1 + geo_W1[h*3+2]*rg2;
      g1f[ck][e] = (__bf16)fmaxf(a, 0.f);
    }

  const float* qrow = q_t + ((size_t)b * NPTS + n) * CDIM;
  const float* krow = k_t + ((size_t)b * NPTS + idxj) * CDIM;
  v16bf relf[4], geof[4];
#pragma unroll
  for (int mt = 0; mt < 8; ++mt) {
    v8f acc = {};
    acc = wmma_bf16(fragA_pk(pGeoW2, mt * 2 + 0, lane), g1f[0], acc);
    acc = wmma_bf16(fragA_pk(pGeoW2, mt * 2 + 1, lane), g1f[1], acc);
    int ck = mt >> 1, eo = (mt & 1) * 8, c0 = mt * 16 + mb;
#pragma unroll
    for (int r = 0; r < 8; ++r) {
      float ge = acc[r] + geo_b2[c0 + r];
      geof[ck][eo + r] = (__bf16)ge;
      relf[ck][eo + r] = (__bf16)(qrow[c0 + r] - krow[c0 + r] + ge);
    }
  }

  v16bf hf[2];
#pragma unroll
  for (int ht = 0; ht < 4; ++ht) {
    v8f acc = {};
#pragma unroll
    for (int kc = 0; kc < 4; ++kc)
      acc = wmma_bf16(fragA_pk(pRelW1, ht * 4 + kc, lane), relf[kc], acc);
    int ck = ht >> 1, eo = (ht & 1) * 8, c0 = ht * 16 + mb;
#pragma unroll
    for (int r = 0; r < 8; ++r)
      hf[ck][eo + r] = (__bf16)fmaxf(acc[r] * bn_scale[c0 + r] + bn_shift[c0 + r], 0.f);
  }

  const float* frow = feats_t + ((size_t)b * NPTS + n) * CDIM;
  const float* vrow = v_t + ((size_t)b * NPTS + idxj) * CDIM;
  float* arow = agg_t + ((size_t)b * NPTS + n) * CDIM;
#pragma unroll
  for (int mt = 0; mt < 8; ++mt) {
    v8f acc = {};
    acc = wmma_bf16(fragA_pk(pRelW2, mt * 2 + 0, lane), hf[0], acc);
    acc = wmma_bf16(fragA_pk(pRelW2, mt * 2 + 1, lane), hf[1], acc);
    int ck = mt >> 1, eo = (mt & 1) * 8, c0 = mt * 16 + mb;
    float rs[8];
#pragma unroll
    for (int r = 0; r < 8; ++r) {
      float lg = acc[r] + rel_b2[c0 + r];
      float m = lg;
#pragma unroll
      for (int s = 1; s < 16; s <<= 1) m = fmaxf(m, __shfl_xor(m, s, 32));
      float e = __expf(lg - m);
      float sum = e;
#pragma unroll
      for (int s = 1; s < 16; s <<= 1) sum += __shfl_xor(sum, s, 32);
      float t = (e / sum) * (vrow[c0 + r] + (float)geof[ck][eo + r]);
#pragma unroll
      for (int s = 1; s < 16; s <<= 1) t += __shfl_xor(t, s, 32);
      rs[r] = t;
    }
    if (jj == 0) {
#pragma unroll
      for (int r = 0; r < 8; ++r)
        arow[c0 + r] = rs[r] + frow[c0 + r];
    }
  }
}

// ---------------------------------------------------------------- out MLP
// grid (N/16, B), block 256. B frags straight from agg_t; hidden tiles shared
// across waves via fragment-major LDS (16B stores / 32B loads per lane).
__global__ void out_mlp_kernel(const float* __restrict__ agg_t,
                               const __bf16* __restrict__ pW1, const float* b1,
                               const __bf16* __restrict__ pW2, const float* b2,
                               float* __restrict__ out) {
  __shared__ __bf16 s_hid[8 * 32 * 16];   // 8 k-chunks of 256-ch hidden, frag layout
  int lane = threadIdx.x & 31, wv = threadIdx.x >> 5;
  int n0 = blockIdx.x * 16, b = blockIdx.y;
  int jj = lane & 15, mb = (lane & 16) ? 8 : 0;

  const float* arow = agg_t + ((size_t)b * NPTS + n0 + jj) * CDIM;
  v16bf bfA[4];
#pragma unroll
  for (int kc = 0; kc < 4; ++kc) bfA[kc] = fragB_rowf32(arow + kc * 32, lane);

#pragma unroll
  for (int half = 0; half < 2; ++half) {
    int mt = wv + half * 8;               // 16 tiles over 256 hidden channels
    v8f acc = {};
#pragma unroll
    for (int kc = 0; kc < 4; ++kc)
      acc = wmma_bf16(fragA_pk(pW1, mt * 4 + kc, lane), bfA[kc], acc);
    union { __bf16 h[8]; float4 f4; } u;
#pragma unroll
    for (int r = 0; r < 8; ++r)
      u.h[r] = (__bf16)fmaxf(acc[r] + b1[mt * 16 + mb + r], 0.f);
    *(float4*)&s_hid[(((mt >> 1) * 32) + lane) * 16 + (mt & 1) * 8] = u.f4;
  }
  __syncthreads();

  v8f acc = {};
#pragma unroll
  for (int kc = 0; kc < 8; ++kc) {
    v16bf hb = *(const v16bf*)&s_hid[(kc * 32 + lane) * 16];
    acc = wmma_bf16(fragA_pk(pW2, wv * 8 + kc, lane), hb, acc);
  }
  float* ob = out + (size_t)b * CDIM * NPTS + n0 + jj;
#pragma unroll
  for (int r = 0; r < 8; ++r) {
    int c = wv * 16 + mb + r;
    ob[(size_t)c * NPTS] = acc[r] + b2[c] + arow[c];
  }
}

// ------------------------------------------------------------------ launcher
extern "C" void kernel_launch(void* const* d_in, const int* in_sizes, int n_in,
                              void* d_out, int out_size, void* d_ws, size_t ws_size,
                              hipStream_t stream) {
  const float* pts    = (const float*)d_in[0];
  const float* feats  = (const float*)d_in[1];
  const float* Wq = (const float*)d_in[2];  const float* bq = (const float*)d_in[3];
  const float* Wk = (const float*)d_in[4];  const float* bk = (const float*)d_in[5];
  const float* Wv = (const float*)d_in[6];  const float* bv = (const float*)d_in[7];
  const float* geo_W1 = (const float*)d_in[8];  const float* geo_b1 = (const float*)d_in[9];
  const float* geo_W2 = (const float*)d_in[10]; const float* geo_b2 = (const float*)d_in[11];
  const float* rel_W1 = (const float*)d_in[12]; /* rel_b1 (d_in[13]) cancels in BN */
  const float* rel_g    = (const float*)d_in[14];
  const float* rel_beta = (const float*)d_in[15];
  const float* rel_W2 = (const float*)d_in[16]; const float* rel_b2 = (const float*)d_in[17];
  const float* out_W1 = (const float*)d_in[18]; const float* out_b1 = (const float*)d_in[19];
  const float* out_W2 = (const float*)d_in[20]; const float* out_b2 = (const float*)d_in[21];

  float* ws = (float*)d_ws;
  const size_t SZ = (size_t)BDIM * CDIM * NPTS;           // 4,194,304 floats
  float* q_t     = ws;
  float* k_t     = ws + SZ;
  float* v_t     = ws + 2 * SZ;
  float* feats_t = ws + 3 * SZ;
  float* agg_t   = ws + 4 * SZ;
  float* stat    = ws + 5 * SZ;                            // 256 floats
  int*   knn     = (int*)(ws + 5 * SZ + 256);              // B*N*16 ints
  __bf16* pw     = (__bf16*)(ws + 5 * SZ + 256 + (size_t)BDIM * NPTS * KNN);
  __bf16* pWq     = pw;
  __bf16* pWk     = pw + 16384;
  __bf16* pWv     = pw + 32768;
  __bf16* pGeoW2  = pw + 49152;
  __bf16* pRelW1  = pw + 57344;
  __bf16* pRelW2  = pw + 65536;
  __bf16* pOutW1  = pw + 73728;
  __bf16* pOutW2  = pw + 106496;

  zero_stats<<<1, 256, 0, stream>>>(stat);
  pack_weights<<<64, 256, 0, stream>>>(Wq,     pWq,    CDIM, CDIM);
  pack_weights<<<64, 256, 0, stream>>>(Wk,     pWk,    CDIM, CDIM);
  pack_weights<<<64, 256, 0, stream>>>(Wv,     pWv,    CDIM, CDIM);
  pack_weights<<<64, 256, 0, stream>>>(geo_W2, pGeoW2, CDIM, HDIM);
  pack_weights<<<64, 256, 0, stream>>>(rel_W1, pRelW1, HDIM, CDIM);
  pack_weights<<<64, 256, 0, stream>>>(rel_W2, pRelW2, CDIM, HDIM);
  pack_weights<<<64, 256, 0, stream>>>(out_W1, pOutW1, 2*CDIM, CDIM);
  pack_weights<<<64, 256, 0, stream>>>(out_W2, pOutW2, CDIM, 2*CDIM);

  transpose_bcn<<<dim3(NPTS / 64, BDIM), 256, 0, stream>>>(feats, feats_t);
  qkv_kernel<<<dim3(NPTS / 16, BDIM, 3), 256, 0, stream>>>(
      feats_t, pWq, bq, pWk, bk, pWv, bv, q_t, k_t, v_t);
  knn_kernel<<<dim3(NPTS / 256, BDIM), 256, 0, stream>>>(pts, knn);
  stats_kernel<<<(BDIM * NPTS) / 8, 256, 0, stream>>>(
      pts, q_t, k_t, knn, geo_W1, geo_b1, pGeoW2, geo_b2, pRelW1,
      stat, stat + 64);
  bn_finalize<<<1, 64, 0, stream>>>(stat, stat + 64, rel_g, rel_beta,
                                    stat + 128, stat + 192);
  attn_agg_kernel<<<(BDIM * NPTS) / 8, 256, 0, stream>>>(
      pts, feats_t, q_t, k_t, v_t, knn, geo_W1, geo_b1, pGeoW2, geo_b2,
      pRelW1, pRelW2, rel_b2, stat + 128, stat + 192, agg_t);
  out_mlp_kernel<<<dim3(NPTS / 16, BDIM), 256, 0, stream>>>(
      agg_t, pOutW1, out_b1, pOutW2, out_b2, (float*)d_out);
}
